// GraphSAGEModel_45389214384862
// MI455X (gfx1250) — matrix operand order repaired and checked by reading, not compiled, not run on hardware
//
#include <hip/hip_runtime.h>

#define NN 50000   // nodes
#define NE 800000  // edges
#define DF 128     // feature dim

typedef __attribute__((ext_vector_type(16))) _Float16     v16h;
typedef __attribute__((ext_vector_type(8)))  float        v8f;
typedef __attribute__((ext_vector_type(4)))  float        f4;
typedef __attribute__((ext_vector_type(4)))  unsigned int u4;

// ---------------------------------------------------------------- utilities
__global__ void zero_f32(float* __restrict__ p, int n) {
  int i = blockIdx.x * blockDim.x + threadIdx.x;
  if (i < n) p[i] = 0.0f;
}

__global__ void degree_kernel(const int* __restrict__ dst, float* __restrict__ deg) {
  int e = blockIdx.x * blockDim.x + threadIdx.x;
  if (e < NE) atomicAdd(&deg[dst[e]], 1.0f);
}

__global__ void cvt_f16(const float* __restrict__ w, _Float16* __restrict__ wh, int n) {
  int i = blockIdx.x * blockDim.x + threadIdx.x;
  if (i < n) wh[i] = (_Float16)w[i];
}

// ------------------------------------------------------- edge scatter (mean-sum)
// One wave32 per edge; lane covers 4 consecutive floats -> coalesced 512B/edge.
// summed[dst] += feat[src]; atomics stay resident in the 192MB L2.
__global__ void scatter_kernel(const float* __restrict__ feat,
                               const int* __restrict__ src,
                               const int* __restrict__ dst,
                               float* __restrict__ summed) {
  int gid  = blockIdx.x * blockDim.x + threadIdx.x;
  int e    = gid >> 5;
  int lane = gid & 31;
  if (e >= NE) return;
  int s = src[e], d = dst[e];
  f4 v = *(const f4*)(feat + (size_t)s * DF + lane * 4);
  float* o = summed + (size_t)d * DF + lane * 4;
  atomicAdd(o + 0, v.x);
  atomicAdd(o + 1, v.y);
  atomicAdd(o + 2, v.z);
  atomicAdd(o + 3, v.w);
}

// ----------------------------------------------- fused SAGE linear via WMMA
// out = act( (summed/deg) @ Wl^T + b + xroot @ Wr^T )
// Block = 128 threads (4 waves). Each wave: 16 nodes x 128 outputs
// = 8 accumulator tiles of v_wmma_f32_16x16x32_f16, K-loop of 4 chunks.
template <bool RELU>
__global__ void __launch_bounds__(128) sage_gemm_kernel(
    const float* __restrict__ summed, const float* __restrict__ deg,
    const float* __restrict__ xroot,
    const _Float16* __restrict__ Wl, const _Float16* __restrict__ Wr,
    const float* __restrict__ bias, float* __restrict__ out) {
  __shared__ _Float16 sh_agg[4][16][DF];  // per-wave staging, pre-scaled f16
  __shared__ _Float16 sh_x[4][16][DF];

  const int wave  = threadIdx.x >> 5;
  const int lane  = threadIdx.x & 31;
  const int node0 = blockIdx.x * 64 + wave * 16;
  const int row   = lane & 15;
  const int hi    = lane >> 4;

  // Stage 16 nodes x 128 feats: one f32x4 per lane per row, scale + cvt f16.
  for (int r = 0; r < 16; ++r) {
    int node = node0 + r;
    f4 sv = (f4)0.0f, xv = (f4)0.0f;
    float inv = 0.0f;
    if (node < NN) {
      inv = 1.0f / fmaxf(deg[node], 1.0f);
      sv  = *(const f4*)(summed + (size_t)node * DF + lane * 4);
      xv  = *(const f4*)(xroot  + (size_t)node * DF + lane * 4);
    }
    union { _Float16 h[4]; unsigned long long u; } pa, px;
    pa.h[0] = (_Float16)(sv.x * inv); pa.h[1] = (_Float16)(sv.y * inv);
    pa.h[2] = (_Float16)(sv.z * inv); pa.h[3] = (_Float16)(sv.w * inv);
    px.h[0] = (_Float16)xv.x;         px.h[1] = (_Float16)xv.y;
    px.h[2] = (_Float16)xv.z;         px.h[3] = (_Float16)xv.w;
    *(unsigned long long*)&sh_agg[wave][r][lane * 4] = pa.u;
    *(unsigned long long*)&sh_x[wave][r][lane * 4]   = px.u;
  }
  __syncthreads();

  v8f acc[8];
#pragma unroll
  for (int t = 0; t < 8; ++t) acc[t] = (v8f)0.0f;

#pragma unroll
  for (int kc = 0; kc < 4; ++kc) {
    // A 16x32 f16, contiguous k order chosen here (B matches): lane<16 -> K {kb..kb+15}, kb = kc*32;
    // lane>=16 -> K {kb+16..kb+31}.
    const int kbase = kc * 32 + hi * 16;
    union { v16h v; u4 u[2]; } aA, aX;
    const _Float16* pa = &sh_agg[wave][row][kbase];
    const _Float16* px = &sh_x[wave][row][kbase];
    aA.u[0] = *(const u4*)(pa);
    aA.u[1] = *(const u4*)(pa + 8);
    aX.u[0] = *(const u4*)(px);
    aX.u[1] = *(const u4*)(px + 8);

#pragma unroll
    for (int ot = 0; ot < 8; ++ot) {
      const int ocol = ot * 16 + row;           // this lane's output column
      const int koff = kc * 32 + hi * 16;       // B 32x16: lane<16 K0..15, lane>=16 K16..31
      v16h bl = *(const v16h*)(Wl + ocol * DF + koff);
      v16h br = *(const v16h*)(Wr + ocol * DF + koff);
      acc[ot] = __builtin_amdgcn_wmma_f32_16x16x32_f16(
          false, aA.v, false, bl, (short)0, acc[ot], false, false);
      acc[ot] = __builtin_amdgcn_wmma_f32_16x16x32_f16(
          false, aX.v, false, br, (short)0, acc[ot], false, false);
    }
  }

  // D layout: VGPR v, lanes 0-15 -> M=v, lanes 16-31 -> M=v+8; N = lane&15.
  const int m0 = hi * 8;
#pragma unroll
  for (int ot = 0; ot < 8; ++ot) {
    const int ocol = ot * 16 + row;
    const float bv = bias[ocol];
#pragma unroll
    for (int v = 0; v < 8; ++v) {
      int node = node0 + m0 + v;
      if (node < NN) {
        float r = acc[ot][v] + bv;
        if (RELU) r = fmaxf(r, 0.0f);
        out[(size_t)node * DF + ocol] = r;
      }
    }
  }
}

// ---------------------------------------------------------------- launcher
extern "C" void kernel_launch(void* const* d_in, const int* in_sizes, int n_in,
                              void* d_out, int out_size, void* d_ws, size_t ws_size,
                              hipStream_t stream) {
  const float* x   = (const float*)d_in[0];
  const int*   ei  = (const int*)d_in[1];  // edge_index [2][E]
  const float* Wl0 = (const float*)d_in[2];
  const float* Wr0 = (const float*)d_in[3];
  const float* b0  = (const float*)d_in[4];
  const float* Wl1 = (const float*)d_in[5];
  const float* Wr1 = (const float*)d_in[6];
  const float* b1  = (const float*)d_in[7];
  float* out = (float*)d_out;

  const int* src = ei;
  const int* dst = ei + NE;

  // workspace layout (floats): summed[N*D] | deg[N] | h[N*D] | f16 weights x4
  float* summed = (float*)d_ws;
  float* deg    = summed + (size_t)NN * DF;
  float* h      = deg + NN;
  _Float16* wh  = (_Float16*)(h + (size_t)NN * DF);  // 32B-aligned by construction
  _Float16* Wl0h = wh + 0 * DF * DF;
  _Float16* Wr0h = wh + 1 * DF * DF;
  _Float16* Wl1h = wh + 2 * DF * DF;
  _Float16* Wr1h = wh + 3 * DF * DF;

  const int nz0 = NN * DF + NN;  // summed + deg are contiguous
  zero_f32<<<(nz0 + 255) / 256, 256, 0, stream>>>(summed, nz0);
  degree_kernel<<<(NE + 255) / 256, 256, 0, stream>>>(dst, deg);

  const int dd = DF * DF;
  cvt_f16<<<(dd + 255) / 256, 256, 0, stream>>>(Wl0, Wl0h, dd);
  cvt_f16<<<(dd + 255) / 256, 256, 0, stream>>>(Wr0, Wr0h, dd);
  cvt_f16<<<(dd + 255) / 256, 256, 0, stream>>>(Wl1, Wl1h, dd);
  cvt_f16<<<(dd + 255) / 256, 256, 0, stream>>>(Wr1, Wr1h, dd);

  const int scatter_blocks = (NE * 32) / 256;  // one wave32 per edge
  const int gemm_blocks    = (NN + 63) / 64;

  // Layer 0: scatter x -> summed; fused (mean, Wl0, b0, Wr0, ReLU) -> h
  scatter_kernel<<<scatter_blocks, 256, 0, stream>>>(x, src, dst, summed);
  sage_gemm_kernel<true><<<gemm_blocks, 128, 0, stream>>>(
      summed, deg, x, Wl0h, Wr0h, b0, h);

  // Layer 1: re-zero summed; scatter h; fused (mean, Wl1, b1, Wr1) -> out
  zero_f32<<<(NN * DF + 255) / 256, 256, 0, stream>>>(summed, NN * DF);
  scatter_kernel<<<scatter_blocks, 256, 0, stream>>>(h, src, dst, summed);
  sage_gemm_kernel<false><<<gemm_blocks, 128, 0, stream>>>(
      summed, deg, h, Wl1h, Wr1h, b1, out);
}
